// LSTM_2181843386433
// MI455X (gfx1250) — compile-verified, matrix operand order checked by reading, not run on hardware
//
#include <hip/hip_runtime.h>
#include <hip/hip_bf16.h>

typedef _Float16 f16;
typedef __attribute__((ext_vector_type(16))) _Float16 v16h;
typedef __attribute__((ext_vector_type(8)))  float    v8f;

#define Bsz   64
#define Tt    512
#define DIMc  512
#define LAT   1024
#define KTOT  (DIMc + LAT)      // 1536
#define KSTEPS (KTOT / 32)      // 48
#define KX    (DIMc / 32)       // 16 k-steps from x, rest from h
#define NT_TOT 256              // 4096/16 column tiles
#define BL    (Bsz * LAT)       // 65536
#define NBLK  64                // persistent grid: 64 blocks * 4 waves = 256 waves

// ---- workspace layout (bytes) ----
#define WS_CNT    0
#define WS_WPK    256
#define WPK_BYTES ((size_t)NT_TOT * KSTEPS * 32 * 32)   // 12,582,912
#define WS_X16    (WS_WPK + WPK_BYTES)
#define X16_BYTES ((size_t)Tt * Bsz * DIMc * 2)         // 33,554,432
#define WS_H16    (WS_X16 + X16_BYTES)
#define H16_BYTES ((size_t)2 * BL * 2)                  // double buffered f16 h
#define WS_CST    (WS_H16 + H16_BYTES)                  // f32 cell state (256 KB)

// Fast transcendentals: v_exp_f32 + v_rcp_f32 (TRANS ops, co-execute with VALU)
__device__ __forceinline__ float frcp(float x) {
    return __builtin_amdgcn_rcpf(x);
}
__device__ __forceinline__ float fsig(float x) {
    return frcp(1.0f + __expf(-x));
}
__device__ __forceinline__ float ftanh(float x) {
    float ax = fabsf(x);
    float e  = __expf(-2.0f * ax);
    float r  = (1.0f - e) * frcp(1.0f + e);
    return copysignf(r, x);
}

// ---- zero h (both buffers), C state, barrier counter ----
__global__ void lstm_init(f16* __restrict__ h16, float* __restrict__ Cst,
                          unsigned* __restrict__ counter) {
    int i = blockIdx.x * blockDim.x + threadIdx.x;
    if (i < 2 * BL) h16[i] = (f16)0.0f;
    if (i < BL)     Cst[i] = 0.0f;
    if (i == 0)     *counter = 0u;
}

// ---- x [B,T,D] f32 -> x16 [T,B,D] f16 (time-major for the scan) ----
__global__ void lstm_cvt_x(const float* __restrict__ x, f16* __restrict__ x16) {
    size_t i = (size_t)blockIdx.x * blockDim.x + threadIdx.x;
    if (i >= (size_t)Bsz * Tt * DIMc) return;
    int    k = (int)(i % DIMc);
    size_t r = i / DIMc;
    int    t = (int)(r % Tt);
    int    b = (int)(r / Tt);
    x16[((size_t)t * Bsz + b) * DIMc + k] = (f16)x[i];
}

// ---- pack W = [Wi;Wf;Wo;Wg] (each [L, D+L] f32) into per-lane WMMA B-fragment
// layout: wpk[((nt*KSTEPS)+s)*32 + lane] = v16h of W[n][k..] per ISA 16-bit
// 16x32 layout (lane&15 selects n within tile; lane>>4 selects k-half).
__global__ void lstm_pack_w(const float* __restrict__ Wi, const float* __restrict__ Wf,
                            const float* __restrict__ Wo, const float* __restrict__ Wg,
                            v16h* __restrict__ wpk) {
    unsigned tid = blockIdx.x * blockDim.x + threadIdx.x;
    if (tid >= (unsigned)(NT_TOT * KSTEPS * 32)) return;
    int lane = tid & 31;
    int s    = (tid >> 5) % KSTEPS;
    int nt   = tid / (32 * KSTEPS);
    int nlo  = lane & 15;
    int koff = (lane >> 4) * 8;
    int ng   = nt * 16 + nlo;                 // global column in [0,4096)
    const float* Wsrc = (ng < 1024) ? Wi : (ng < 2048) ? Wf : (ng < 3072) ? Wo : Wg;
    const float* rp   = Wsrc + (size_t)(ng & 1023) * KTOT;
    v16h o;
#pragma unroll
    for (int v = 0; v < 8; ++v) {
        int kb = s * 32 + ((v & 3) << 1) + ((v >> 2) << 4) + koff;
        o[2 * v]     = (f16)rp[kb];
        o[2 * v + 1] = (f16)rp[kb + 1];
    }
    wpk[tid] = o;
}

// ---- persistent scan kernel: 64 blocks x 128 threads (256 waves).
// wave w: mTile = w&3 (batch tile), jTile = w>>2 (per-gate column tile).
// Each wave computes z tiles for all 4 gates at (mTile, jTile) and fuses the
// LSTM cell update; one atomic grid barrier per time step.
__global__ __launch_bounds__(128) void lstm_scan(
    const f16* __restrict__ x16, const v16h* __restrict__ wpk,
    f16* __restrict__ h16, float* __restrict__ Cst,
    unsigned* __restrict__ counter,
    const float* __restrict__ bi, const float* __restrict__ bfv,
    const float* __restrict__ bo, const float* __restrict__ bg,
    float* __restrict__ out) {
    const int lane  = threadIdx.x & 31;
    const int w     = blockIdx.x * 4 + (threadIdx.x >> 5);
    const int mTile = w & 3;
    const int jTile = w >> 2;
    const int mlo   = lane & 15;
    const int koff  = (lane >> 4) * 8;
    const int m     = mTile * 16 + mlo;       // batch row this lane feeds into A
    const int j     = jTile * 16 + mlo;       // per-gate output column (C/D lane)
    const float bv0 = bi[j], bv1 = bfv[j], bv2 = bo[j], bv3 = bg[j];

    const v16h* wp0 = wpk + ((size_t)(0 * 64 + jTile) * KSTEPS) * 32 + lane;
    const v16h* wp1 = wpk + ((size_t)(1 * 64 + jTile) * KSTEPS) * 32 + lane;
    const v16h* wp2 = wpk + ((size_t)(2 * 64 + jTile) * KSTEPS) * 32 + lane;
    const v16h* wp3 = wpk + ((size_t)(3 * 64 + jTile) * KSTEPS) * 32 + lane;

    for (int t = 0; t < Tt; ++t) {
        const f16* xrow = x16 + ((size_t)t * Bsz + m) * DIMc;
        const f16* hrow = h16 + (size_t)(t & 1) * BL + (size_t)m * LAT;
        v8f acc0 = {}, acc1 = {}, acc2 = {}, acc3 = {};

        for (int s = 0; s < KSTEPS; ++s) {
            // A fragment: 16x32 f16, lane holds row m, k-pairs per ISA layout
            // (pairs are contiguous -> compiler merges into 2x global_load_b128)
            const f16* src;
            int kb0;
            if (s < KX) { src = xrow; kb0 = s * 32 + koff; }
            else        { src = hrow; kb0 = (s - KX) * 32 + koff; }
            v16h a;
#pragma unroll
            for (int v = 0; v < 8; ++v) {
                int kb = kb0 + ((v & 3) << 1) + ((v >> 2) << 4);
                a[2 * v]     = src[kb];
                a[2 * v + 1] = src[kb + 1];
            }
            // B fragments (pre-packed, contiguous 32B/lane)
            v16h b0 = wp0[(size_t)s * 32];
            v16h b1 = wp1[(size_t)s * 32];
            v16h b2 = wp2[(size_t)s * 32];
            v16h b3 = wp3[(size_t)s * 32];
            acc0 = __builtin_amdgcn_wmma_f32_16x16x32_f16(false, a, false, b0, (short)0, acc0, false, false);
            acc1 = __builtin_amdgcn_wmma_f32_16x16x32_f16(false, a, false, b1, (short)0, acc1, false, false);
            acc2 = __builtin_amdgcn_wmma_f32_16x16x32_f16(false, a, false, b2, (short)0, acc2, false, false);
            acc3 = __builtin_amdgcn_wmma_f32_16x16x32_f16(false, a, false, b3, (short)0, acc3, false, false);
        }

        // fused LSTM cell epilogue; C/D layout: VGPR v, lane -> row v+8*(lane>>4)
        f16* hnext = h16 + (size_t)((t + 1) & 1) * BL;
#pragma unroll
        for (int v = 0; v < 8; ++v) {
            int    mb  = mTile * 16 + v + ((lane >> 4) << 3);
            size_t idx = (size_t)mb * LAT + j;
            float zi = acc0[v] + bv0;
            float zf = acc1[v] + bv1;
            float zo = acc2[v] + bv2;
            float zg = acc3[v] + bv3;
            float Cn = fsig(zf) * Cst[idx] + fsig(zi) * ftanh(zg);
            float h  = fsig(zo) * ftanh(Cn);
            Cst[idx] = Cn;
            out[(size_t)t * BL + idx] = h;            // h_hist[t]
            hnext[idx] = (f16)h;
            if (t == Tt - 1) out[(size_t)Tt * BL + idx] = Cn;  // C_final
        }

        // warm the caches for the next step's x row while we wait at the barrier
        if (t + 1 < Tt) {
            const char* nx = (const char*)(x16 + ((size_t)(t + 1) * Bsz + m) * DIMc);
            __builtin_prefetch(nx, 0, 1);
            __builtin_prefetch(nx + 512, 0, 1);       // row is 1 KB
        }

        // grid-wide barrier between time steps (monotonic generation counter)
        __syncthreads();
        if (threadIdx.x == 0) {
            __threadfence();
            atomicAdd(counter, 1u);
            unsigned target = (unsigned)(NBLK * (t + 1));
            while (__hip_atomic_load(counter, __ATOMIC_RELAXED, __HIP_MEMORY_SCOPE_AGENT) < target) {
                __builtin_amdgcn_s_sleep(1);
            }
            __threadfence();
        }
        __syncthreads();
    }
}

extern "C" void kernel_launch(void* const* d_in, const int* in_sizes, int n_in,
                              void* d_out, int out_size, void* d_ws, size_t ws_size,
                              hipStream_t stream) {
    const float* x  = (const float*)d_in[0];
    const float* Wi = (const float*)d_in[1];
    const float* bi = (const float*)d_in[2];
    const float* Wf = (const float*)d_in[3];
    const float* bf = (const float*)d_in[4];
    const float* Wo = (const float*)d_in[5];
    const float* bo = (const float*)d_in[6];
    const float* Wg = (const float*)d_in[7];
    const float* bg = (const float*)d_in[8];
    float* out = (float*)d_out;

    char*     ws      = (char*)d_ws;
    unsigned* counter = (unsigned*)(ws + WS_CNT);
    v16h*     wpk     = (v16h*)(ws + WS_WPK);
    f16*      x16     = (f16*)(ws + WS_X16);
    f16*      h16     = (f16*)(ws + WS_H16);
    float*    Cst     = (float*)(ws + WS_CST);

    lstm_init<<<(2 * BL + 255) / 256, 256, 0, stream>>>(h16, Cst, counter);
    lstm_cvt_x<<<(int)(((size_t)Bsz * Tt * DIMc + 255) / 256), 256, 0, stream>>>(x, x16);
    lstm_pack_w<<<(NT_TOT * KSTEPS * 32 + 255) / 256, 256, 0, stream>>>(Wi, Wf, Wo, Wg, wpk);
    lstm_scan<<<NBLK, 128, 0, stream>>>(x16, wpk, h16, Cst, counter,
                                        bi, bf, bo, bg, out);
}
